// HDSymbolicAttention_43087111913943
// MI455X (gfx1250) — matrix operand order repaired and checked by reading, not compiled, not run on hardware
//
#include <hip/hip_runtime.h>

// ---------------------------------------------------------------------------
// HDSymbolicAttention for MI455X (gfx1250, wave32, WMMA)
//
// Pipeline:
//   k_tobf16_pad : f32 -> bf16 with +/-1 zero guard row per batch (values, symbols)
//   k_tobf16     : f32 -> bf16 (conv1_w, conv2_w)
//   k_conv1      : conv1(k5,s3,SAME) bf16 WMMA GEMM (K=2560), branch-free inner
//                  loop via padded input, software-pipelined; +bias+LN(128)+SiLU
//   k_conv2      : conv2(k3,s5,SAME) bf16 WMMA GEMM (K=384, N=1024); +bias+LN+SiLU
//   k_scores     : S = mean(sign(a)*sign(a+sp)) + softmax
//   k_attn       : att = scores@vp via f32 16x16x4 WMMA; O = silu(att*sp)
// ---------------------------------------------------------------------------

typedef __attribute__((ext_vector_type(16))) __bf16 v16bf;
typedef __attribute__((ext_vector_type(8)))  float  v8f;
typedef __attribute__((ext_vector_type(2)))  float  v2f;

#define B_   4
#define T_   1920
#define TP_  1922   // padded rows: index i in [-1, 1920] stored at row i+1
#define E_   512
#define T1_  640
#define C1_  128
#define T2_  128
#define C2_  1024
#define G_   5      // 4 value batches + 1 symbol batch

// ---- workspace layout (bytes) ----
#define OFF_VALSP  ((size_t)0)                                  // 4*1922*512 bf16
#define OFF_SYMSP  (OFF_VALSP + (size_t)B_*TP_*E_*2)            // 1922*512 bf16
#define OFF_W1     (OFF_SYMSP + (size_t)TP_*E_*2)               // 5*512*128 bf16
#define OFF_W2     (OFF_W1 + (size_t)5*E_*C1_*2)                // 3*128*1024 bf16
#define OFF_X1     (OFF_W2 + (size_t)3*C1_*C2_*2)               // 5*640*128 bf16
#define OFF_X2     (OFF_X1 + (size_t)G_*T1_*C1_*2)              // 5*128*1024 f32
#define OFF_SCORES (OFF_X2 + (size_t)G_*T2_*C2_*4)              // 4*128*128 f32

__device__ __forceinline__ unsigned short f2bf(float f) {
    unsigned u = __builtin_bit_cast(unsigned, f);
    unsigned r = u + 0x7FFFu + ((u >> 16) & 1u);   // round-to-nearest-even
    return (unsigned short)(r >> 16);
}

__device__ __forceinline__ float silu(float x) {
    return x / (1.f + __expf(-x));
}

// ---------------------------------------------------------------------------
__global__ __launch_bounds__(256) void k_tobf16(const float* __restrict__ in,
                                                unsigned short* __restrict__ out,
                                                int n) {
    int i = blockIdx.x * blockDim.x + threadIdx.x;
    if (i < n) out[i] = f2bf(in[i]);
}

// writes element i of batch g at padded offset i + (2g+1)*E_  (one guard row
// before each batch, one after)
__global__ __launch_bounds__(256) void k_tobf16_pad(const float* __restrict__ in,
                                                    unsigned short* __restrict__ out,
                                                    int n, int batch_elems) {
    int i = blockIdx.x * blockDim.x + threadIdx.x;
    if (i < n) {
        int g = i / batch_elems;
        out[(size_t)i + (size_t)(2 * g + 1) * E_] = f2bf(in[i]);
    }
}

// zero the 10 guard rows (2 per padded batch: 4 values + 1 symbols)
__global__ __launch_bounds__(256) void k_zeropad(unsigned short* __restrict__ vals_pad,
                                                 unsigned short* __restrict__ syms_pad) {
    int r = blockIdx.x;               // 0..9
    unsigned short* p;
    if (r < 8) {
        int g = r >> 1;
        p = vals_pad + (size_t)g * TP_ * E_ + (size_t)((r & 1) ? (TP_ - 1) : 0) * E_;
    } else {
        p = syms_pad + (size_t)((r & 1) ? (TP_ - 1) : 0) * E_;
    }
    for (int i = threadIdx.x; i < E_; i += blockDim.x) p[i] = 0;
}

// ---------------------------------------------------------------------------
// conv1: out[g, t, n] = sum_{kw,e} x[g, 3t-1+kw, e] * w1[kw, e, n]   (K = 2560)
// block = 256 thr (8 waves); block covers 16 output rows x 128 cols.
// Input is zero-padded, so the K-loop is branch-free; software-pipelined.
__global__ __launch_bounds__(256) void k_conv1(const unsigned short* __restrict__ vals_pad,
                                               const unsigned short* __restrict__ syms_pad,
                                               const unsigned short* __restrict__ w1_bf,
                                               const float* __restrict__ b1,
                                               const float* __restrict__ g1,
                                               const float* __restrict__ bl1,
                                               unsigned short* __restrict__ X1) {
    const int blk  = blockIdx.x;            // 0..199
    const int g    = blk / 40;
    const int t0   = (blk % 40) * 16;
    const int tid  = threadIdx.x;
    const int wave = tid >> 5;
    const int lane = tid & 31;
    const int half = lane >> 4;
    const int m    = lane & 15;
    const int n    = (wave << 4) + m;       // output channel 0..127

    // base such that row index i (in [-1,1920]) addresses src + i*E_
    const unsigned short* src =
        ((g < 4) ? (vals_pad + (size_t)g * TP_ * E_) : syms_pad) + E_;

    // A row start for this lane (row m of the M-tile), i = 3t-1+kw
    const unsigned short* arow = src + (ptrdiff_t)((t0 + m) * 3 - 1) * E_ + (half << 3);
    // B: global K = 32*s + kk;  lane -> column n, kk in table order (8*half + ...)
    const unsigned short* brow =
        w1_bf + (size_t)(half << 3) * C1_ + (wave << 4) + m;

    auto loadA = [&](int s) -> v16bf {
        const int kw = s >> 4;
        const int e0 = (s & 15) << 5;
        const unsigned short* p = arow + (ptrdiff_t)kw * E_ + e0;
        v16bf a;
        ((uint4*)&a)[0] = *(const uint4*)p;
        ((uint4*)&a)[1] = *(const uint4*)(p + 16);
        return a;
    };
    auto loadB = [&](int s) -> v16bf {
        const unsigned short* q = brow + (size_t)s * 32 * C1_;
        v16bf bm;
#pragma unroll
        for (int i = 0; i < 16; ++i)
            bm[i] = __builtin_bit_cast(__bf16, q[(size_t)(i < 8 ? i : i + 8) * C1_]);
        return bm;
    };

    v8f acc = {};
    v16bf a = loadA(0);
    v16bf bm = loadB(0);
#pragma unroll 2
    for (int s = 0; s < 79; ++s) {
        __builtin_prefetch(brow + (size_t)(s + 2) * 32 * C1_, 0, 3);
        v16bf an = loadA(s + 1);          // issue next fragments first ...
        v16bf bn = loadB(s + 1);
        acc = __builtin_amdgcn_wmma_f32_16x16x32_bf16(false, a, false, bm,
                                                      (short)0, acc, false, false);
        a = an;                            // ... so the WMMA only waits on old loads
        bm = bn;
    }
    acc = __builtin_amdgcn_wmma_f32_16x16x32_bf16(false, a, false, bm,
                                                  (short)0, acc, false, false);

    // bias + LayerNorm(128) + SiLU, all from registers via LDS float atomics
    __shared__ float lsum[16], lsq[16], lmean[16], lrstd[16];
    if (tid < 16) { lsum[tid] = 0.f; lsq[tid] = 0.f; }
    __syncthreads();

    const float bias = b1[n];
    float x[8];
#pragma unroll
    for (int r = 0; r < 8; ++r) {
        x[r] = acc[r] + bias;
        const int row = r + (half << 3);
        atomicAdd(&lsum[row], x[r]);
        atomicAdd(&lsq[row], x[r] * x[r]);
    }
    __syncthreads();
    if (tid < 16) {
        const float mean = lsum[tid] * (1.f / 128.f);
        const float var  = lsq[tid] * (1.f / 128.f) - mean * mean;
        lmean[tid] = mean;
        lrstd[tid] = rsqrtf(var + 1e-3f);
    }
    __syncthreads();
    const float gg = g1[n], bb = bl1[n];
#pragma unroll
    for (int r = 0; r < 8; ++r) {
        const int row = r + (half << 3);
        float y = (x[r] - lmean[row]) * lrstd[row] * gg + bb;
        y = silu(y);
        X1[((size_t)g * T1_ + t0 + row) * C1_ + n] = f2bf(y);
    }
}

// ---------------------------------------------------------------------------
// conv2: out[g, t, n] = sum_{kw,c} X1[g, 5t+kw, c] * w2[kw, c, n]   (K = 384, N = 1024)
// block = 512 thr (16 waves); block covers 16 rows x 1024 cols; wave -> 4 N-tiles.
// i = 5t+kw <= 637 < 640, so no padding is needed.
__global__ __launch_bounds__(512) void k_conv2(const unsigned short* __restrict__ X1,
                                               const unsigned short* __restrict__ w2_bf,
                                               const float* __restrict__ b2,
                                               const float* __restrict__ g2,
                                               const float* __restrict__ bl2,
                                               float* __restrict__ X2,
                                               float* __restrict__ vp_out) {
    const int blk  = blockIdx.x;            // 0..39
    const int g    = blk / 8;
    const int t0   = (blk % 8) * 16;
    const int tid  = threadIdx.x;
    const int wave = tid >> 5;
    const int lane = tid & 31;
    const int half = lane >> 4;
    const int m    = lane & 15;

    const unsigned short* arow =
        X1 + (size_t)g * T1_ * C1_ + (size_t)(t0 + m) * 5 * C1_ + (half << 3);
    // global K = 32*s + kk
    const unsigned short* brow =
        w2_bf + (size_t)(half << 3) * C2_ + m;

    auto loadA = [&](int s) -> v16bf {
        const int kw = s >> 2;
        const int c0 = (s & 3) << 5;
        const unsigned short* p = arow + (size_t)kw * C1_ + c0;
        v16bf a;
        ((uint4*)&a)[0] = *(const uint4*)p;
        ((uint4*)&a)[1] = *(const uint4*)(p + 16);
        return a;
    };

    v8f acc[4] = {{}, {}, {}, {}};
    v16bf a = loadA(0);
#pragma unroll 2
    for (int s = 0; s < 12; ++s) {
        v16bf an;
        if (s < 11) an = loadA(s + 1);     // pipeline the shared A fragment
        const unsigned short* q0 = brow + (size_t)s * 32 * C2_;
        __builtin_prefetch(q0 + 32 * C2_, 0, 3);
#pragma unroll
        for (int qi = 0; qi < 4; ++qi) {
            const int n0 = ((wave << 2) + qi) << 4;
            v16bf bm;
#pragma unroll
            for (int i = 0; i < 16; ++i)
                bm[i] = __builtin_bit_cast(__bf16, q0[n0 + (size_t)(i < 8 ? i : i + 8) * C2_]);
            acc[qi] = __builtin_amdgcn_wmma_f32_16x16x32_bf16(false, a, false, bm,
                                                              (short)0, acc[qi], false, false);
        }
        a = an;
    }

    __shared__ float lsum[16], lsq[16], lmean[16], lrstd[16];
    if (tid < 16) { lsum[tid] = 0.f; lsq[tid] = 0.f; }
    __syncthreads();

    float x[4][8];
#pragma unroll
    for (int r = 0; r < 8; ++r) {
        float ps = 0.f, ps2 = 0.f;
#pragma unroll
        for (int qi = 0; qi < 4; ++qi) {
            const int n = ((((wave << 2) + qi) << 4)) + m;
            const float v = acc[qi][r] + b2[n];
            x[qi][r] = v;
            ps += v;
            ps2 += v * v;
        }
        const int row = r + (half << 3);
        atomicAdd(&lsum[row], ps);
        atomicAdd(&lsq[row], ps2);
    }
    __syncthreads();
    if (tid < 16) {
        const float mean = lsum[tid] * (1.f / 1024.f);
        const float var  = lsq[tid] * (1.f / 1024.f) - mean * mean;
        lmean[tid] = mean;
        lrstd[tid] = rsqrtf(var + 1e-3f);
    }
    __syncthreads();
#pragma unroll
    for (int r = 0; r < 8; ++r) {
        const int row = r + (half << 3);
        const int t = t0 + row;
#pragma unroll
        for (int qi = 0; qi < 4; ++qi) {
            const int n = ((((wave << 2) + qi) << 4)) + m;
            float y = (x[qi][r] - lmean[row]) * lrstd[row] * g2[n] + bl2[n];
            y = silu(y);
            const size_t o = ((size_t)g * T2_ + t) * C2_ + n;
            X2[o] = y;
            if (g < 4) vp_out[o] = y;   // values_projected output (second tuple element)
        }
    }
}

// ---------------------------------------------------------------------------
// S[b,j,i] = (1/1024) * sum_d sign(vp[b,i,d]) * sign(vp[b,i,d] + sp[j,d]); softmax over i.
// block = 256 thr (8 waves) per (b, group of 8 j's); wave w owns j = j0+w.
__global__ __launch_bounds__(256) void k_scores(const float* __restrict__ X2,
                                                float* __restrict__ scores) {
    const int b    = blockIdx.x >> 4;
    const int j0   = (blockIdx.x & 15) * 8;
    const int tid  = threadIdx.x;
    const int wave = tid >> 5;
    const int lane = tid & 31;

    __shared__ float spl[8 * C2_];
    __shared__ float vprow[C2_];
    __shared__ float Srow[8][T2_];

    const float* sp = X2 + (size_t)4 * T2_ * C2_;
    const float* vp = X2 + (size_t)b * T2_ * C2_;

    for (int idx = tid; idx < 8 * C2_; idx += 256)
        spl[idx] = sp[(size_t)(j0 + (idx >> 10)) * C2_ + (idx & (C2_ - 1))];
    __syncthreads();

    const float* sj = &spl[wave << 10];
    for (int i = 0; i < T2_; ++i) {
        for (int idx = tid; idx < C2_; idx += 256)
            vprow[idx] = vp[(size_t)i * C2_ + idx];
        __syncthreads();
        float s = 0.f;
#pragma unroll 8
        for (int d = lane; d < C2_; d += 32) {
            const float a  = vprow[d];
            const float ab = a + sj[d];
            const float sa = (float)((a > 0.f) - (a < 0.f));
            const float sb = (float)((ab > 0.f) - (ab < 0.f));
            s += sa * sb;
        }
        for (int o = 16; o > 0; o >>= 1) s += __shfl_xor(s, o, 32);
        if (lane == 0) Srow[wave][i] = s * (1.f / 1024.f);
        __syncthreads();
    }

    // softmax over i (wave w handles row j0+w; 4 elements per lane)
    float v[4], mx = -1e30f;
#pragma unroll
    for (int k = 0; k < 4; ++k) {
        v[k] = Srow[wave][lane + 32 * k];
        mx = fmaxf(mx, v[k]);
    }
    for (int o = 16; o > 0; o >>= 1) mx = fmaxf(mx, __shfl_xor(mx, o, 32));
    float sum = 0.f;
#pragma unroll
    for (int k = 0; k < 4; ++k) { v[k] = __expf(v[k] - mx); sum += v[k]; }
    for (int o = 16; o > 0; o >>= 1) sum += __shfl_xor(sum, o, 32);
    const float inv = 1.f / sum;
#pragma unroll
    for (int k = 0; k < 4; ++k)
        scores[((size_t)b * T2_ + j0 + wave) * T2_ + lane + 32 * k] = v[k] * inv;
}

// ---------------------------------------------------------------------------
// att[b,j,d] = sum_i scores[b,j,i] * vp[b,i,d]  via f32 16x16x4 WMMA (exact fp32);
// O = silu(att * sp).   block = 256 thr (8 waves); wave -> one 16x16 (j,d) tile.
__global__ __launch_bounds__(256) void k_attn(const float* __restrict__ scores,
                                              const float* __restrict__ X2,
                                              float* __restrict__ O) {
    const int blk  = blockIdx.x;            // 0..255 : b(4) x jt(8) x dg(8)
    const int b    = blk >> 6;
    const int jt   = (blk >> 3) & 7;
    const int dg   = blk & 7;
    const int tid  = threadIdx.x;
    const int wave = tid >> 5;
    const int lane = tid & 31;
    const int half = lane >> 4;
    const int m    = lane & 15;
    const int j0   = jt << 4;
    const int d0   = ((dg << 3) + wave) << 4;

    // A (16x4 f32): lanes<16 -> K={0,1}, lanes>=16 -> K={2,3}
    const float* ap = scores + (size_t)b * T2_ * T2_ + (size_t)(j0 + m) * T2_ + (half << 1);
    // B (4x16 f32): mirror layout, lane -> N=m
    const float* bp = X2 + (size_t)b * T2_ * C2_ + (size_t)(half << 1) * C2_ + d0 + m;

    v8f acc = {};
#pragma unroll 4
    for (int s = 0; s < 32; ++s) {
        const int k0 = s << 2;
        v2f a = *(const v2f*)(ap + k0);
        v2f bm;
        bm.x = bp[(size_t)k0 * C2_];
        bm.y = bp[(size_t)(k0 + 1) * C2_];
        acc = __builtin_amdgcn_wmma_f32_16x16x4_f32(false, a, false, bm,
                                                    (short)0, acc, false, false);
    }

    const float* sp = X2 + (size_t)4 * T2_ * C2_;
#pragma unroll
    for (int r = 0; r < 8; ++r) {
        const int j = j0 + r + (half << 3);
        const int d = d0 + m;
        const float xv = acc[r] * sp[(size_t)j * C2_ + d];
        O[((size_t)b * T2_ + j) * C2_ + d] = silu(xv);
    }
}

// ---------------------------------------------------------------------------
extern "C" void kernel_launch(void* const* d_in, const int* in_sizes, int n_in,
                              void* d_out, int out_size, void* d_ws, size_t ws_size,
                              hipStream_t stream) {
    const float* values  = (const float*)d_in[0];
    const float* symbols = (const float*)d_in[1];
    const float* conv1_w = (const float*)d_in[2];
    const float* conv1_b = (const float*)d_in[3];
    const float* ln1_g   = (const float*)d_in[4];
    const float* ln1_b   = (const float*)d_in[5];
    const float* conv2_w = (const float*)d_in[6];
    const float* conv2_b = (const float*)d_in[7];
    const float* ln2_g   = (const float*)d_in[8];
    const float* ln2_b   = (const float*)d_in[9];

    char* ws = (char*)d_ws;
    unsigned short* vals_pad = (unsigned short*)(ws + OFF_VALSP);
    unsigned short* syms_pad = (unsigned short*)(ws + OFF_SYMSP);
    unsigned short* w1_bf    = (unsigned short*)(ws + OFF_W1);
    unsigned short* w2_bf    = (unsigned short*)(ws + OFF_W2);
    unsigned short* X1       = (unsigned short*)(ws + OFF_X1);
    float*          X2       = (float*)(ws + OFF_X2);
    float*          scores   = (float*)(ws + OFF_SCORES);

    float* O      = (float*)d_out;
    float* vp_out = O + (size_t)B_ * T2_ * C2_;   // values_projected (tuple element 2)

    // 1) guard rows + bf16 conversions
    k_zeropad<<<10, 256, 0, stream>>>(vals_pad, syms_pad);
    {
        const int nv = B_ * T_ * E_;
        k_tobf16_pad<<<(nv + 255) / 256, 256, 0, stream>>>(values, vals_pad, nv, T_ * E_);
        const int ns = T_ * E_;
        k_tobf16_pad<<<(ns + 255) / 256, 256, 0, stream>>>(symbols, syms_pad, ns, ns);
        const int n1 = 5 * E_ * C1_;
        k_tobf16<<<(n1 + 255) / 256, 256, 0, stream>>>(conv1_w, w1_bf, n1);
        const int n2 = 3 * C1_ * C2_;
        k_tobf16<<<(n2 + 255) / 256, 256, 0, stream>>>(conv2_w, w2_bf, n2);
    }
    // 2) conv1 + LN + SiLU  (5 "batches" x 40 row-tiles)
    k_conv1<<<G_ * 40, 256, 0, stream>>>(vals_pad, syms_pad, w1_bf, conv1_b, ln1_g, ln1_b, X1);
    // 3) conv2 + LN + SiLU  (writes values_projected to d_out too)
    k_conv2<<<G_ * 8, 512, 0, stream>>>(X1, w2_bf, conv2_b, ln2_g, ln2_b, X2, vp_out);
    // 4) sign-similarity + softmax
    k_scores<<<B_ * 16, 256, 0, stream>>>(X2, scores);
    // 5) attention GEMM (f32 WMMA) + silu epilogue
    k_attn<<<B_ * 64, 256, 0, stream>>>(scores, X2, O);
}
